// Model_80221399155460
// MI455X (gfx1250) — compile-verified, matrix operand order checked by reading, not run on hardware
//
#include <hip/hip_runtime.h>

// ---------------------------------------------------------------------------
// Types
// ---------------------------------------------------------------------------
typedef __bf16 bf16_t;
typedef __attribute__((ext_vector_type(16))) __bf16 v16bf;
typedef __attribute__((ext_vector_type(8)))  __bf16 v8bf;
typedef __attribute__((ext_vector_type(8)))  float  v8f;
typedef __attribute__((ext_vector_type(4)))  int    v4i;

// ---------------------------------------------------------------------------
// CDNA5 async global->LDS 16B copy (ASYNCcnt) with sync fallback.
// Builtin signature (probe-confirmed via diagnostic): (v4i AS1*, v4i AS3*, Ii, Ii)
// ---------------------------------------------------------------------------
#if defined(__has_builtin)
#if __has_builtin(__builtin_amdgcn_global_load_async_to_lds_b128)
#define HAS_ASYNC_LDS 1
#endif
#endif
#ifndef HAS_ASYNC_LDS
#define HAS_ASYNC_LDS 0
#endif

__device__ __forceinline__ void cp16(bf16_t* lds_dst, const bf16_t* gsrc) {
#if HAS_ASYNC_LDS
  __builtin_amdgcn_global_load_async_to_lds_b128(
      (__attribute__((address_space(1))) v4i*)gsrc,
      (__attribute__((address_space(3))) v4i*)lds_dst, 0, 0);
#else
  *(v8bf*)lds_dst = *(const v8bf*)gsrc;
#endif
}

__device__ __forceinline__ void cp_wait() {
#if HAS_ASYNC_LDS
#if __has_builtin(__builtin_amdgcn_s_wait_asynccnt)
  __builtin_amdgcn_s_wait_asynccnt(0);
#else
  asm volatile("s_wait_asynccnt 0" ::: "memory");
#endif
#endif
}

// ---------------------------------------------------------------------------
// WMMA 16x16x32 bf16 fragment loader.
// Lane L holds row (L&15); K chunks [kh*8,+8) and [16+kh*8,+8), kh=L>>4.
// Both are 16B-contiguous -> two ds_load_b128 from row-major LDS.
// ---------------------------------------------------------------------------
__device__ __forceinline__ v16bf load_frag(const bf16_t* rowbase, int kh) {
  const v8bf lo = *(const v8bf*)(rowbase + kh * 8);
  const v8bf hi = *(const v8bf*)(rowbase + 16 + kh * 8);
  v16bf f;
#pragma unroll
  for (int i = 0; i < 8; ++i) { f[i] = lo[i]; f[8 + i] = hi[i]; }
  return f;
}

__device__ __forceinline__ float activate(float v, int act) {
  if (act == 1) return v > 0.f ? v : 0.f;
  if (act == 2) return 1.f / (1.f + __expf(-v));
  return v;
}

#define WMMA_BF16(a, b, c) \
  __builtin_amdgcn_wmma_f32_16x16x32_bf16(false, (a), false, (b), (short)0, (c), false, false)

// ---------------------------------------------------------------------------
// NT GEMM: C[M,N] = act(A[M,Kp] * B[Np,Kp]^T + bias)
// A,B bf16 row-major with padded pitch ldab (multiple of 32, zero pads).
// M % 128 == 0. Grid.x covers Np/64. Block 128x64, 8 waves -> 32x32 each.
// Double-buffered LDS tiles filled with async b128 copies.
// ---------------------------------------------------------------------------
#define BM 128
#define BN 64
#define BK 32
#define SPITCH 40  // 32 bf16 + 8 pad -> 80B rows, 16B aligned

__global__ __launch_bounds__(256) void gemm_bf16_wmma(
    const bf16_t* __restrict__ A, const bf16_t* __restrict__ Bw,
    const float* __restrict__ bias, float* __restrict__ Cf,
    bf16_t* __restrict__ Cb, int N, int ldab, int ldc, int act)
{
  __shared__ __attribute__((aligned(16))) bf16_t sA[2][BM * SPITCH];
  __shared__ __attribute__((aligned(16))) bf16_t sB[2][BN * SPITCH];

  const int tid  = threadIdx.x;
  const int lane = tid & 31;
  const int wave = tid >> 5;
  const int bm   = blockIdx.y * BM;
  const int bn   = blockIdx.x * BN;
  const int wm   = (wave >> 1) * 32;
  const int wn   = (wave & 1) * 32;
  const int kh   = lane >> 4;
  const int r16  = lane & 15;
  const int nk   = ldab >> 5;  // zero K-pads are inert

  v8f acc00 = {}, acc01 = {}, acc10 = {}, acc11 = {};

  auto stage = [&](int kt, int p) {
    const int k0 = kt * BK;
    // A: 128 rows x 4 chunks of 16B = 512 chunks (2 per thread)
    {
      int idx = tid;
      int row = idx >> 2, c = (idx & 3) * 8;
      cp16(&sA[p][row * SPITCH + c], A + (size_t)(bm + row) * ldab + k0 + c);
      idx += 256;
      row = idx >> 2; c = (idx & 3) * 8;
      cp16(&sA[p][row * SPITCH + c], A + (size_t)(bm + row) * ldab + k0 + c);
    }
    // B: 64 rows x 4 chunks = 256 chunks (1 per thread)
    {
      const int row = tid >> 2, c = (tid & 3) * 8;
      cp16(&sB[p][row * SPITCH + c], Bw + (size_t)(bn + row) * ldab + k0 + c);
    }
  };

  stage(0, 0);
  for (int kt = 0; kt < nk; ++kt) {
    const int p = kt & 1;
    cp_wait();
    __syncthreads();
    if (kt + 1 < nk) stage(kt + 1, p ^ 1);

    const bf16_t* aB = &sA[p][0];
    const bf16_t* bB = &sB[p][0];
    const v16bf a0 = load_frag(aB + (wm + r16) * SPITCH, kh);
    const v16bf a1 = load_frag(aB + (wm + 16 + r16) * SPITCH, kh);
    const v16bf b0 = load_frag(bB + (wn + r16) * SPITCH, kh);
    const v16bf b1 = load_frag(bB + (wn + 16 + r16) * SPITCH, kh);
    acc00 = WMMA_BF16(a0, b0, acc00);
    acc01 = WMMA_BF16(a0, b1, acc01);
    acc10 = WMMA_BF16(a1, b0, acc10);
    acc11 = WMMA_BF16(a1, b1, acc11);
  }

  // epilogue: VGPR r -> M = r + 8*kh; lane&15 -> N
  auto emit = [&](int m, int n, float v) {
    if (n < N) {
      v += bias ? bias[n] : 0.f;
      v = activate(v, act);
    } else {
      v = 0.f;  // zero the K-pad columns of the next layer's operand
    }
    if (Cf && n < N) Cf[(size_t)m * N + n] = v;
    if (Cb && n < ldc) Cb[(size_t)m * ldc + n] = (bf16_t)v;
  };
#pragma unroll
  for (int r = 0; r < 8; ++r) {
    const int m0 = bm + wm + r + kh * 8;
    const int n0 = bn + wn + r16;
    emit(m0,      n0,      acc00[r]);
    emit(m0,      n0 + 16, acc01[r]);
    emit(m0 + 16, n0,      acc10[r]);
    emit(m0 + 16, n0 + 16, acc11[r]);
  }
}

// ---------------------------------------------------------------------------
// Persistent GRU recurrence. 25 blocks, each owns 16 H-columns; 48 gate rows
// of w_hh (bf16, pitch 416) pinned in LDS for all T steps. Per step: async-
// stage h_t (bf16 ping-pong, pitch 416) in double-buffered 32-wide K chunks,
// 13 WMMAs per compute wave, fp32 gate math, device-wide atomic barrier.
// ---------------------------------------------------------------------------
#define GRU_H 400
#define GRU_B 32
#define GRU_KP 416
#define WPITCH 424  // 848B rows, 16B aligned

__global__ __launch_bounds__(256) void gru_persistent(
    const float* __restrict__ xp,      // [B*T, 3H] row = b*T + t (incl. b_ih)
    const bf16_t* __restrict__ whhb,   // [3H, 416] bf16, zero K-pads
    const float* __restrict__ bhh,     // [3H]
    float* __restrict__ hping,         // [2][B*400] fp32 state
    bf16_t* __restrict__ hb16,         // [2][B*416] bf16 state (pads zero)
    bf16_t* __restrict__ yout,         // [B*T, 416] bf16 outputs
    float* __restrict__ hlast,         // [B*400] final hidden (d_out)
    unsigned* __restrict__ bar,        // [0]=count [1]=generation
    int T)
{
  __shared__ __attribute__((aligned(16))) bf16_t sW[48 * WPITCH];
  __shared__ __attribute__((aligned(16))) bf16_t sH[2][GRU_B * 40];
  __shared__ float sGH[48 * 32];
  __shared__ float sBias[48];

  const int tid  = threadIdx.x;
  const int lane = tid & 31;
  const int wave = tid >> 5;
  const int c0   = blockIdx.x * 16;
  const int kh   = lane >> 4;
  const int r16  = lane & 15;
  const int mt   = wave & 1;   // batch half (waves 0..5 compute)
  const int gt   = wave >> 1;  // gate r/z/n

  // pin w_hh slice: 48 rows x 26 chunks of 16B = 1248 chunks
  for (int idx = tid; idx < 48 * 26; idx += 256) {
    const int rowL = idx / 26;
    const int col  = (idx % 26) * 8;
    const int grow = (rowL >> 4) * GRU_H + c0 + (rowL & 15);
    cp16(&sW[rowL * WPITCH + col], whhb + (size_t)grow * GRU_KP + col);
  }
  if (tid < 48) sBias[tid] = bhh[(tid >> 4) * GRU_H + c0 + (tid & 15)];
  cp_wait();
  __syncthreads();

  for (int t = 0; t < T; ++t) {
    const float*  hcur   = hping + (t & 1) * (GRU_B * GRU_H);
    float*        hnext  = hping + ((t + 1) & 1) * (GRU_B * GRU_H);
    const bf16_t* hbcur  = hb16 + (t & 1) * (GRU_B * GRU_KP);
    bf16_t*       hbnext = hb16 + ((t + 1) & 1) * (GRU_B * GRU_KP);

    // stage h chunk: 32 rows x 4 chunks of 16B = 128 chunks
    auto stageH = [&](int kt, int p) {
      if (tid < 128) {
        const int row = tid >> 2, c = (tid & 3) * 8;
        cp16(&sH[p][row * 40 + c], hbcur + (size_t)row * GRU_KP + kt * 32 + c);
      }
    };
    stageH(0, 0);

    v8f acc = {};
    for (int kt = 0; kt < 13; ++kt) {
      const int p = kt & 1;
      cp_wait();
      __syncthreads();
      if (kt + 1 < 13) stageH(kt + 1, p ^ 1);
      if (wave < 6) {
        const v16bf a = load_frag(&sH[p][(mt * 16 + r16) * 40], kh);
        const v16bf b = load_frag(sW + (gt * 16 + r16) * WPITCH + kt * 32, kh);
        acc = WMMA_BF16(a, b, acc);
      }
    }
    __syncthreads();

    if (wave < 6) {
#pragma unroll
      for (int r = 0; r < 8; ++r) {
        const int m = mt * 16 + r + kh * 8;
        sGH[(gt * 16 + r16) * 32 + m] = acc[r] + sBias[gt * 16 + r16];
      }
    }
    __syncthreads();

    // gate math: 512 (b, j) pairs, 2 per thread
    for (int i = tid; i < GRU_B * 16; i += 256) {
      const int b   = i >> 4;
      const int j   = i & 15;
      const int col = c0 + j;
      const size_t xb = ((size_t)b * T + t) * (3 * GRU_H);
      const float gr = sGH[j * 32 + b];
      const float gz = sGH[(16 + j) * 32 + b];
      const float gn = sGH[(32 + j) * 32 + b];
      const float r_ = 1.f / (1.f + __expf(-(xp[xb + col] + gr)));
      const float z_ = 1.f / (1.f + __expf(-(xp[xb + GRU_H + col] + gz)));
      const float n_ = tanhf(xp[xb + 2 * GRU_H + col] + r_ * gn);
      const float ho = hcur[b * GRU_H + col];
      const float hn = (1.f - z_) * n_ + z_ * ho;
      hnext[b * GRU_H + col] = hn;
      hbnext[b * GRU_KP + col] = (bf16_t)hn;
      yout[((size_t)b * T + t) * GRU_KP + col] = (bf16_t)hn;
      if (t == T - 1) hlast[b * GRU_H + col] = hn;
    }

    // device-wide step barrier (atomic count + generation, s_sleep spin)
    __syncthreads();
    if (tid == 0) {
      __threadfence();
      const unsigned g = __hip_atomic_load(&bar[1], __ATOMIC_RELAXED,
                                           __HIP_MEMORY_SCOPE_AGENT);
      const unsigned arrived = __hip_atomic_fetch_add(
          &bar[0], 1u, __ATOMIC_ACQ_REL, __HIP_MEMORY_SCOPE_AGENT);
      if (arrived == gridDim.x - 1) {
        __hip_atomic_store(&bar[0], 0u, __ATOMIC_RELAXED,
                           __HIP_MEMORY_SCOPE_AGENT);
        __hip_atomic_store(&bar[1], g + 1u, __ATOMIC_RELEASE,
                           __HIP_MEMORY_SCOPE_AGENT);
      } else {
        while (__hip_atomic_load(&bar[1], __ATOMIC_ACQUIRE,
                                 __HIP_MEMORY_SCOPE_AGENT) == g) {
          __builtin_amdgcn_s_sleep(2);
        }
      }
    }
    __syncthreads();
  }
}

// ---------------------------------------------------------------------------
// Helpers
// ---------------------------------------------------------------------------
__global__ void k_zero_u32(unsigned* p, int n) {
  const int i = blockIdx.x * blockDim.x + threadIdx.x;
  if (i < n) p[i] = 0u;
}
__global__ void k_fill_zero16(uint4* __restrict__ p, size_t n16) {
  uint4 z; z.x = z.y = z.z = z.w = 0u;
  for (size_t i = (size_t)blockIdx.x * blockDim.x + threadIdx.x; i < n16;
       i += (size_t)gridDim.x * blockDim.x)
    p[i] = z;
}
// fp32 [rows,cols] -> bf16 [rowsP,colsP], zero pads
__global__ void k_cvt_pad(const float* __restrict__ s, bf16_t* __restrict__ d,
                          int rows, int cols, int rowsP, int colsP) {
  const size_t total = (size_t)rowsP * colsP;
  for (size_t i = (size_t)blockIdx.x * blockDim.x + threadIdx.x; i < total;
       i += (size_t)gridDim.x * blockDim.x) {
    const int r = (int)(i / colsP);
    const int c = (int)(i % colsP);
    d[i] = (r < rows && c < cols) ? (bf16_t)s[(size_t)r * cols + c]
                                  : (bf16_t)0.0f;
  }
}
// initialize one layer's hidden state (fp32 ping-pong + padded bf16 copy)
__global__ void k_init_h(const float* __restrict__ h0, float* __restrict__ hp,
                         bf16_t* __restrict__ hb) {
  const int i = blockIdx.x * blockDim.x + threadIdx.x;
  if (i < 2 * GRU_B * GRU_KP) {
    const int buf = i / (GRU_B * GRU_KP);
    const int r   = (i / GRU_KP) % GRU_B;
    const int c   = i % GRU_KP;
    const float v = (buf == 0 && c < GRU_H) ? h0[r * GRU_H + c] : 0.f;
    hb[i] = (bf16_t)v;
    if (c < GRU_H) hp[buf * GRU_B * GRU_H + r * GRU_H + c] = v;
  }
}

// ---------------------------------------------------------------------------
// Host orchestration
// ---------------------------------------------------------------------------
extern "C" void kernel_launch(void* const* d_in, const int* in_sizes, int n_in,
                              void* d_out, int out_size, void* d_ws,
                              size_t ws_size, hipStream_t stream) {
  (void)in_sizes; (void)n_in; (void)out_size; (void)ws_size;
  const int B = 32, T = 1000, I = 257, Hh = 400;
  const int M = B * T;  // 32000, multiple of 128
  // padded K pitches (multiples of 32) and padded N row counts (mult. of 64)
  const int KpI = 288, KpH = 416, Kp6 = 608;
  const int NpH = 448, Np6 = 640, Np3H = 1216, NpI = 320;

  const float* x      = (const float*)d_in[0];
  const float* h1     = (const float*)d_in[1];
  const float* h2     = (const float*)d_in[2];
  const float* fc1_w  = (const float*)d_in[3];
  const float* fc1_b  = (const float*)d_in[4];
  const float* g1_wih = (const float*)d_in[5];
  const float* g1_whh = (const float*)d_in[6];
  const float* g1_bih = (const float*)d_in[7];
  const float* g1_bhh = (const float*)d_in[8];
  const float* g2_wih = (const float*)d_in[9];
  const float* g2_whh = (const float*)d_in[10];
  const float* g2_bih = (const float*)d_in[11];
  const float* g2_bhh = (const float*)d_in[12];
  const float* fc3_w  = (const float*)d_in[13];
  const float* fc3_b  = (const float*)d_in[14];
  const float* fc4_w  = (const float*)d_in[15];
  const float* fc4_b  = (const float*)d_in[16];
  const float* fc2_w  = (const float*)d_in[17];
  const float* fc2_b  = (const float*)d_in[18];

  char* ws = (char*)d_ws;
  size_t off = 0;
  auto carve = [&](size_t bytes) -> char* {
    char* p = ws + off;
    off = (off + bytes + 255) & ~(size_t)255;
    return p;
  };
  unsigned* bar  = (unsigned*)carve(256);
  float* h1p     = (float*)carve(2 * B * Hh * sizeof(float));
  float* h2p     = (float*)carve(2 * B * Hh * sizeof(float));
  bf16_t* h1b16  = (bf16_t*)carve(2 * B * KpH * 2);
  bf16_t* h2b16  = (bf16_t*)carve(2 * B * KpH * 2);
  bf16_t* xb     = (bf16_t*)carve((size_t)M * KpI * 2);
  bf16_t* fc1wb  = (bf16_t*)carve((size_t)NpH * KpI * 2);
  bf16_t* wih1b  = (bf16_t*)carve((size_t)Np3H * KpH * 2);
  bf16_t* whh1b  = (bf16_t*)carve((size_t)3 * Hh * KpH * 2);
  bf16_t* wih2b  = (bf16_t*)carve((size_t)Np3H * KpH * 2);
  bf16_t* whh2b  = (bf16_t*)carve((size_t)3 * Hh * KpH * 2);
  bf16_t* fc3wb  = (bf16_t*)carve((size_t)Np6 * KpH * 2);
  bf16_t* fc4wb  = (bf16_t*)carve((size_t)Np6 * Kp6 * 2);
  bf16_t* fc2wb  = (bf16_t*)carve((size_t)NpI * Kp6 * 2);
  bf16_t* actA   = (bf16_t*)carve((size_t)M * Kp6 * 2);
  bf16_t* actB   = (bf16_t*)carve((size_t)M * Kp6 * 2);
  float*  xpbuf  = (float*)carve((size_t)M * 3 * Hh * sizeof(float));

  float* out_y  = (float*)d_out;             // [B,T,257]
  float* out_h1 = out_y + (size_t)M * I;     // [B,400]
  float* out_h2 = out_h1 + (size_t)B * Hh;   // [B,400]

  // init: barrier, hidden states, actB (its first writer leaves K-pads alone)
  k_zero_u32<<<1, 32, 0, stream>>>(bar, 2);
  k_init_h<<<(2 * B * KpH + 255) / 256, 256, 0, stream>>>(h1, h1p, h1b16);
  k_init_h<<<(2 * B * KpH + 255) / 256, 256, 0, stream>>>(h2, h2p, h2b16);
  k_fill_zero16<<<2048, 256, 0, stream>>>((uint4*)actB,
                                          ((size_t)M * Kp6 * 2) / 16);

  // fp32 -> padded bf16 conversions
  auto cvt = [&](const float* s, bf16_t* d, int r, int c, int rP, int cP) {
    k_cvt_pad<<<2048, 256, 0, stream>>>(s, d, r, c, rP, cP);
  };
  cvt(x, xb, M, I, M, KpI);
  cvt(fc1_w, fc1wb, Hh, I, NpH, KpI);
  cvt(g1_wih, wih1b, 3 * Hh, Hh, Np3H, KpH);
  cvt(g1_whh, whh1b, 3 * Hh, Hh, 3 * Hh, KpH);
  cvt(g2_wih, wih2b, 3 * Hh, Hh, Np3H, KpH);
  cvt(g2_whh, whh2b, 3 * Hh, Hh, 3 * Hh, KpH);
  cvt(fc3_w, fc3wb, 600, Hh, Np6, KpH);
  cvt(fc4_w, fc4wb, 600, 600, Np6, Kp6);
  cvt(fc2_w, fc2wb, I, 600, NpI, Kp6);

  auto gemm = [&](const bf16_t* A, const bf16_t* Bw, const float* bias,
                  float* Cf, bf16_t* Cb, int N, int Npad, int ldab, int ldc,
                  int act) {
    dim3 grid(Npad / BN, M / BM);
    gemm_bf16_wmma<<<grid, 256, 0, stream>>>(A, Bw, bias, Cf, Cb, N, ldab,
                                             ldc, act);
  };

  // fc1: y1 = x*fc1_w^T + b                 -> actA [M,416] bf16
  gemm(xb, fc1wb, fc1_b, nullptr, actA, Hh, NpH, KpI, KpH, 0);
  // xp1 = y1*w_ih1^T + b_ih1                -> xpbuf [M,1200] fp32
  gemm(actA, wih1b, g1_bih, xpbuf, nullptr, 3 * Hh, Np3H, KpH, 0, 0);
  // GRU layer 1 -> y2 in actB (pitch 416), h1_out
  gru_persistent<<<dim3(Hh / 16), 256, 0, stream>>>(
      xpbuf, whh1b, g1_bhh, h1p, h1b16, actB, out_h1, bar, T);
  // xp2 = y2*w_ih2^T + b_ih2
  gemm(actB, wih2b, g2_bih, xpbuf, nullptr, 3 * Hh, Np3H, KpH, 0, 0);
  // GRU layer 2 -> y3 in actA (pitch 416; pads zeroed by fc1 epilogue), h2_out
  gru_persistent<<<dim3(Hh / 16), 256, 0, stream>>>(
      xpbuf, whh2b, g2_bhh, h2p, h2b16, actA, out_h2, bar, T);
  // fc3+relu: -> actB [M,608] bf16 (full pitch rewritten incl. pads)
  gemm(actA, fc3wb, fc3_b, nullptr, actB, 600, Np6, KpH, Kp6, 1);
  // fc4+relu: -> actA [M,608] bf16
  gemm(actB, fc4wb, fc4_b, nullptr, actA, 600, Np6, Kp6, Kp6, 1);
  // fc2+sigmoid: -> d_out [M,257] fp32
  gemm(actA, fc2wb, fc2_b, out_y, nullptr, I, NpI, Kp6, 0, 2);
}
